// HBNS_10651518894821
// MI455X (gfx1250) — matrix operand order repaired and checked by reading, not compiled, not run on hardware
//
#include <hip/hip_runtime.h>
#include <hip/hip_bf16.h>

// Problem constants (match reference).
#define NSRC 12288
#define NTGT 12288
#define CIN  64
#define COUT 32
#define BATCH 2
#define SSTRIPS 16   // s-dimension split for occupancy: 48*2*16 = 1536 blocks

#define TPAD 40      // halves per tile row (80 B: 16B-aligned, bank-friendly)
#define RPAD 34      // floats per red row (lanes 16-31 land 16 banks away)

typedef __attribute__((ext_vector_type(16))) __bf16 v16bf;
typedef __attribute__((ext_vector_type(8)))  float  v8f;
typedef __attribute__((ext_vector_type(2)))  float  vf2;
typedef __attribute__((ext_vector_type(2)))  __bf16 vbf2;

union BF16Vec {
    v16bf v;
    uint4 q[2];           // 32 bytes = 16 bf16
};

// Packed f32x2 -> bf16x2 (lowers to v_cvt_pk_bf16_f32).
__device__ __forceinline__ unsigned int pkbf(float a, float b) {
    vf2 f;
    f[0] = a;
    f[1] = b;
    vbf2 h = __builtin_convertvector(f, vbf2);
    return __builtin_bit_cast(unsigned int, h);
}

__device__ __forceinline__ unsigned short f2bf(float f) {
    return __builtin_bit_cast(unsigned short, (__bf16)f);
}

// ---------------------------------------------------------------------------
// Projection: out[b][d][s] = bf16( sum_c x[b][s][c] * w[c][d] )  (transposed
// [d][s] layout so the aggregation kernel's B-operand loads are contiguous).
// blockIdx.y = b*2 + which (0 = source/w_s, 1 = target/w_t).
// ---------------------------------------------------------------------------
__global__ void proj_bf16_T(const float* __restrict__ xs,
                            const float* __restrict__ xt,
                            const float* __restrict__ ws,
                            const float* __restrict__ wt,
                            unsigned short* __restrict__ smsgT,
                            unsigned short* __restrict__ tmsgT) {
    const int s     = blockIdx.x * blockDim.x + threadIdx.x;
    const int which = blockIdx.y & 1;
    const int b     = blockIdx.y >> 1;
    const float* x   = which ? xt : xs;
    const float* wgt = which ? wt : ws;
    unsigned short* out = which ? tmsgT : smsgT;
    const int N = which ? NTGT : NSRC;
    if (s >= N) return;

    const float* xrow = x + ((size_t)b * N + s) * CIN;
    float xr[CIN];
#pragma unroll
    for (int c = 0; c < CIN; ++c) xr[c] = xrow[c];

    float acc[COUT];
#pragma unroll
    for (int d = 0; d < COUT; ++d) acc[d] = 0.f;
    for (int c = 0; c < CIN; ++c) {
        const float xv = xr[c];
#pragma unroll
        for (int d = 0; d < COUT; ++d) acc[d] += xv * wgt[c * COUT + d];
    }
#pragma unroll
    for (int d = 0; d < COUT; ++d)
        out[((size_t)b * COUT + d) * N + s] = f2bf(acc[d]);
}

// ---------------------------------------------------------------------------
// Fused aggregation. Block = 256 threads = 8 waves, owns 256 t-rows (32/wave)
// and one s-strip of NSRC/SSTRIPS columns; streams s in 32-wide chunks so the
// 1.21 GB neighborhood matrix is read exactly once across the grid.
//   mot[t,d] += N[t,s] * smsg[s,d]  -> register accum, atomic flush per block
//   mos[s,d] += N[t,s] * tmsg[t,d]  -> LDS cross-wave reduce + global atomics
// ---------------------------------------------------------------------------
__launch_bounds__(256)
__global__ void agg_fused(const float* __restrict__ nbhd,
                          const unsigned short* __restrict__ smsgT,
                          const unsigned short* __restrict__ tmsgT,
                          float* __restrict__ mos,
                          float* __restrict__ mot) {
    __shared__ unsigned short tileN[8][32][TPAD]; // [wave][t_local][s_local]
    __shared__ unsigned short tileT[8][32][TPAD]; // [wave][s_local][t_local]
    __shared__ float red[32 * RPAD];              // block partial of mos chunk

    const int tid  = threadIdx.x;
    const int w    = tid >> 5;
    const int lane = tid & 31;
    const int l15  = lane & 15;
    const int hi   = lane >> 4;       // half-wave select
    const int kb   = hi ? 8 : 0;      // A operand: K bases {kb, kb+16}
    const int tb   = hi ? 16 : 0;     // unused (B operand K bases are {kb, kb+16})
    const int b    = blockIdx.y;
    const int t0   = blockIdx.x * 256;
    const int tw   = t0 + w * 32;     // this wave's first t row

    const float* Nb = nbhd + (size_t)b * NTGT * NSRC;

    v8f acc1[2][2];
#pragma unroll
    for (int m = 0; m < 2; ++m)
#pragma unroll
        for (int n = 0; n < 2; ++n) acc1[m][n] = (v8f){};

    // B operand for the mos product (t-message, K = this wave's 32 t rows)
    // is loop-invariant: hoist. Lane reads K {8h..8h+7, 16+8h..16+8h+7}.
    BF16Vec b2[2];
#pragma unroll
    for (int n = 0; n < 2; ++n) {
        const uint4* g = (const uint4*)(tmsgT +
            ((size_t)b * COUT + 16 * n + l15) * NTGT + tw);
        b2[n].q[0] = g[hi];
        b2[n].q[1] = g[2 + hi];
    }

    const int sBeg = blockIdx.z * (NSRC / SSTRIPS);
    const int sEnd = sBeg + (NSRC / SSTRIPS);

    for (int s0 = sBeg; s0 < sEnd; s0 += 32) {
        __syncthreads();                       // prev flush finished reading red
#pragma unroll
        for (int i = 0; i < 4; ++i) {
            const int idx = tid + 256 * i;
            red[(idx >> 5) * RPAD + (idx & 31)] = 0.f;
        }

        // Stage this wave's 32x32 fp32 tile of N -> bf16, both layouts.
        {
            const int r0 = lane >> 3;          // 0..3
            const int c0 = (lane & 7) * 4;     // 0,4,...,28
#pragma unroll
            for (int i = 0; i < 8; ++i) {
                const int r = r0 + i * 4;
                const float4 v =
                    *(const float4*)(Nb + (size_t)(tw + r) * NSRC + s0 + c0);
                const unsigned int p01 = pkbf(v.x, v.y);
                const unsigned int p23 = pkbf(v.z, v.w);
                *(uint2*)&tileN[w][r][c0] = make_uint2(p01, p23);
                union { unsigned int u; unsigned short h[2]; } u0, u1;
                u0.u = p01;
                u1.u = p23;
                tileT[w][c0 + 0][r] = u0.h[0];
                tileT[w][c0 + 1][r] = u0.h[1];
                tileT[w][c0 + 2][r] = u1.h[0];
                tileT[w][c0 + 3][r] = u1.h[1];
            }
        }
        __syncthreads();                       // tiles + red zero visible

        // A operands from LDS (layout matches 16-bit WMMA A striping).
        BF16Vec a1[2], a2[2], b1[2];
#pragma unroll
        for (int m = 0; m < 2; ++m) {
            const uint4* p  = (const uint4*)&tileN[w][16 * m + l15][0];
            a1[m].q[0] = p[kb >> 3];           // K = kb .. kb+7
            a1[m].q[1] = p[(kb + 16) >> 3];    // K = kb+16 .. kb+23
            const uint4* pt = (const uint4*)&tileT[w][16 * m + l15][0];
            a2[m].q[0] = pt[kb >> 3];
            a2[m].q[1] = pt[(kb + 16) >> 3];
        }
        // B operand for mot: s-message, K = this s chunk (table-order load).
#pragma unroll
        for (int n = 0; n < 2; ++n) {
            const uint4* g = (const uint4*)(smsgT +
                ((size_t)b * COUT + 16 * n + l15) * NSRC + s0);
            b1[n].q[0] = g[hi];
            b1[n].q[1] = g[2 + hi];
        }

        // mot accumulation: D += A1(N tile) x B1(s_msg)
#pragma unroll
        for (int m = 0; m < 2; ++m)
#pragma unroll
            for (int n = 0; n < 2; ++n)
                acc1[m][n] = __builtin_amdgcn_wmma_f32_16x16x32_bf16(
                    false, a1[m].v, false, b1[n].v, (short)0, acc1[m][n],
                    false, false);

        // mos partial: D = A2(N^T tile) x B2(t_msg), K = wave's 32 t rows
        v8f d2[2][2];
#pragma unroll
        for (int m = 0; m < 2; ++m)
#pragma unroll
            for (int n = 0; n < 2; ++n) {
                v8f z = {};
                d2[m][n] = __builtin_amdgcn_wmma_f32_16x16x32_bf16(
                    false, a2[m].v, false, b2[n].v, (short)0, z, false, false);
            }

        // Cross-wave reduce the 32x32 mos partial in LDS (ds_add_f32).
#pragma unroll
        for (int m = 0; m < 2; ++m)
#pragma unroll
            for (int n = 0; n < 2; ++n)
#pragma unroll
                for (int r = 0; r < 8; ++r) {
                    const int srow = 16 * m + 8 * hi + r;
                    const int dcol = 16 * n + l15;
                    atomicAdd(&red[srow * RPAD + dcol], d2[m][n][r]);
                }
        __syncthreads();

        // One block-level atomic flush per chunk.
#pragma unroll
        for (int i = 0; i < 4; ++i) {
            const int idx  = tid + 256 * i;
            const int srow = idx >> 5;
            const int dcol = idx & 31;
            unsafeAtomicAdd(
                &mos[((size_t)b * NSRC + s0 + srow) * COUT + dcol],
                red[srow * RPAD + dcol]);
        }
    }

    // Flush mot: SSTRIPS blocks contribute partial sums per t row -> atomics.
#pragma unroll
    for (int m = 0; m < 2; ++m)
#pragma unroll
        for (int n = 0; n < 2; ++n)
#pragma unroll
            for (int r = 0; r < 8; ++r) {
                const int trow = tw + 16 * m + 8 * hi + r;
                const int dcol = 16 * n + l15;
                unsafeAtomicAdd(
                    &mot[((size_t)b * NTGT + trow) * COUT + dcol],
                    acc1[m][n][r]);
            }
}

extern "C" void kernel_launch(void* const* d_in, const int* in_sizes, int n_in,
                              void* d_out, int out_size, void* d_ws, size_t ws_size,
                              hipStream_t stream) {
    const float* xs = (const float*)d_in[0];   // x_source [B,NS,64]
    const float* xt = (const float*)d_in[1];   // x_target [B,NT,64]
    const float* nb = (const float*)d_in[2];   // neighborhood [B,NT,NS]
    const float* ws = (const float*)d_in[3];   // w_s [64,32]
    const float* wt = (const float*)d_in[4];   // w_t [64,32]

    float* mos = (float*)d_out;                          // [B,NS,32]
    float* mot = mos + (size_t)BATCH * NSRC * COUT;      // [B,NT,32]

    unsigned short* smsgT = (unsigned short*)d_ws;                 // [B,32,NS] bf16
    unsigned short* tmsgT = smsgT + (size_t)BATCH * COUT * NSRC;   // [B,32,NT] bf16

    // Both outputs are accumulated with atomics -> zero every launch.
    hipMemsetAsync(d_out, 0,
                   (size_t)BATCH * (NSRC + NTGT) * COUT * sizeof(float),
                   stream);

    dim3 pg(NSRC / 256, BATCH * 2);
    proj_bf16_T<<<pg, 256, 0, stream>>>(xs, xt, ws, wt, smsgT, tmsgT);

    dim3 ag(NTGT / 256, BATCH, SSTRIPS);
    agg_fused<<<ag, 256, 0, stream>>>(nb, smsgT, tmsgT, mos, mot);
}
